// eGAUSSp_65867618451708
// MI455X (gfx1250) — matrix operand order repaired and checked by reading, not compiled, not run on hardware
//
#include <hip/hip_runtime.h>
#include <stdint.h>

// ---------------------------------------------------------------------------
// eGAUSSp on MI455X (gfx1250): Mahalanobis d2 as one folded bf16 WMMA GEMM.
//   d2[b,c] = ZZ_aug[b,:] . Baug[c,:],  K = 1088 (= 34 x 32)
//     k in [0,1024):   ZZ = z_d*z_e          Baug = S_inv[c,d,e]
//     k in [1024,1056) ZZ = z_e              Baug = -2 * (S_inv mu)[c,e]
//     k == 1056        ZZ = 1                Baug = mu' S_inv mu
// ---------------------------------------------------------------------------

typedef __attribute__((ext_vector_type(16))) __bf16 v16bf;
typedef __attribute__((ext_vector_type(8)))  __bf16 v8bf;
typedef __attribute__((ext_vector_type(8)))  float  v8f;

#define BB 65536
#define CC 128
#define DD 32
#define KK 10
#define KTOT 1088          // 1024 + 32 + 1 + 31 zero pad
#define KSTEPS 34          // KTOT / 32
#define ROWS 64            // b-rows per workgroup
#define ROWSTRIDE 1096     // halfs per LDS row (1088 + 8 pad, kills bank conflicts)
#define GSTRIDE 132        // floats per Gamma LDS row

__device__ __forceinline__ unsigned short f2bf(float f) {
    union { float f; uint32_t u; } v; v.f = f;
    const uint32_t u = v.u + 0x7FFFu + ((v.u >> 16) & 1u);   // round-to-nearest-even
    return (unsigned short)(u >> 16);
}

// ---------------------------------------------------------------------------
// Prep: build Baug in WMMA-B fragment-swizzled order (bf16).
// B order used here (A matches): lane L holds N = L%16, 16 consecutive K at
// kbase = s*32 + (L<16 ? 0 : 16). We store each lane's 16 halfs contiguously
// so the main loop's B loads are fully coalesced b128 pairs.
// ---------------------------------------------------------------------------
__global__ void prep_B(const float* __restrict__ S_inv, const float* __restrict__ mu,
                       unsigned short* __restrict__ Bsw) {
    const int idx = blockIdx.x * blockDim.x + threadIdx.x;
    if (idx >= CC * KTOT) return;
    const int c = idx / KTOT;
    const int k = idx - c * KTOT;

    float v = 0.0f;
    if (k < 1024) {
        v = S_inv[c * 1024 + k];
    } else if (k < 1056) {
        const int e = k - 1024;                 // -2 * (S_inv mu)[c][e]
        float s = 0.f;
        for (int d = 0; d < DD; ++d) s += S_inv[c * 1024 + e * DD + d] * mu[c * DD + d];
        v = -2.0f * s;
    } else if (k == 1056) {                     // mu' S_inv mu
        float t3 = 0.f;
        for (int d = 0; d < DD; ++d) {
            float s = 0.f;
            for (int e = 0; e < DD; ++e) s += S_inv[c * 1024 + d * DD + e] * mu[c * DD + e];
            t3 += mu[c * DD + d] * s;
        }
        v = t3;
    }
    const int ct   = c >> 4;
    const int s    = k >> 5;
    const int lane = (c & 15) + (((k & 31) < 16) ? 0 : 16);
    const int slot = k & 15;
    Bsw[(((size_t)(ct * KSTEPS + s) * 32) + lane) * 16 + slot] = f2bf(v);
}

__global__ void prep_L(const int* __restrict__ labels, float* __restrict__ Lmask) {
    const int idx = blockIdx.x * blockDim.x + threadIdx.x;
    if (idx >= KK * CC) return;
    const int k = idx / CC, c = idx - k * CC;
    Lmask[k * CC + c] = (float)labels[c * KK + k];   // labels is [C][K] one-hot
}

// ---------------------------------------------------------------------------
// Main kernel: 256 threads (8 waves). WG = 64 b-rows x all 128 clusters.
// ---------------------------------------------------------------------------
__global__ __launch_bounds__(256)
void egaussp_main(const float* __restrict__ data,
                  const unsigned short* __restrict__ Bsw,
                  const float* __restrict__ Lmask,
                  float* __restrict__ out) {
    extern __shared__ char smem[];
    unsigned short* sA = (unsigned short*)smem;   // ROWS x ROWSTRIDE bf16 (ZZ_aug tile)
    float*          sG = (float*)smem;            // reused: ROWS x GSTRIDE f32 (Gamma)

    const int tid    = threadIdx.x;
    const int lane   = tid & 31;                  // wave32
    const int wave   = tid >> 5;                  // c-tile 0..7
    const int wgBase = blockIdx.x * ROWS;

    // ---- Stage ZZ_aug (bf16) into LDS: 4 threads per row, 8 d-values each ----
    {
        const int r = tid & 63;
        const int q = tid >> 6;
        const float* zrow = data + (size_t)(wgBase + r) * DD;
        float z[DD];
        #pragma unroll
        for (int i = 0; i < DD; i += 4) {
            const float4 t = *(const float4*)(zrow + i);
            z[i] = t.x; z[i + 1] = t.y; z[i + 2] = t.z; z[i + 3] = t.w;
        }
        unsigned short* row = sA + r * ROWSTRIDE;
        #pragma unroll
        for (int dd = 0; dd < 8; ++dd) {
            const int d = q * 8 + dd;
            const float zd = z[d];
            uint32_t p[16];
            #pragma unroll
            for (int e = 0; e < 32; e += 2)
                p[e >> 1] = (uint32_t)f2bf(zd * z[e]) | ((uint32_t)f2bf(zd * z[e + 1]) << 16);
            uint4* dst = (uint4*)(row + d * 32);
            #pragma unroll
            for (int j = 0; j < 4; ++j)
                dst[j] = make_uint4(p[4 * j], p[4 * j + 1], p[4 * j + 2], p[4 * j + 3]);
        }
        if (q == 1) {                              // augmented tail: z, 1.0, zeros
            uint32_t p[32];
            #pragma unroll
            for (int e = 0; e < 32; e += 2)
                p[e >> 1] = (uint32_t)f2bf(z[e]) | ((uint32_t)f2bf(z[e + 1]) << 16);
            p[16] = (uint32_t)f2bf(1.0f);
            #pragma unroll
            for (int j = 17; j < 32; ++j) p[j] = 0u;
            uint4* dst = (uint4*)(row + 1024);
            #pragma unroll
            for (int j = 0; j < 8; ++j)
                dst[j] = make_uint4(p[4 * j], p[4 * j + 1], p[4 * j + 2], p[4 * j + 3]);
        }
    }
    __syncthreads();

    // ---- GEMM: 4 sub-tiles of 16 rows, K = 34 steps of 32, bf16 WMMA ----
    v8f acc[4] = {v8f{}, v8f{}, v8f{}, v8f{}};
    const int rowA = lane & 15;
    const int kSel = (lane >> 4) & 1;             // A-lane K-chunk select
    const unsigned short* bptr = Bsw + ((size_t)wave * KSTEPS * 32 + lane) * 16;

    for (int s = 0; s < KSTEPS; ++s) {
        const v8bf blo = *(const v8bf*)(bptr);
        const v8bf bhi = *(const v8bf*)(bptr + 8);
        bptr += 32 * 16;                          // next k-step (coalesced 1KB/wave)
        const v16bf bfrag = __builtin_shufflevector(blo, bhi,
            0, 1, 2, 3, 4, 5, 6, 7, 8, 9, 10, 11, 12, 13, 14, 15);
        #pragma unroll
        for (int sub = 0; sub < 4; ++sub) {
            const unsigned short* ar =
                sA + (sub * 16 + rowA) * ROWSTRIDE + s * 32 + kSel * 16;
            const v8bf alo = *(const v8bf*)(ar);       // K +0..7
            const v8bf ahi = *(const v8bf*)(ar + 8);   // K +8..15
            const v16bf afrag = __builtin_shufflevector(alo, ahi,
                0, 1, 2, 3, 4, 5, 6, 7, 8, 9, 10, 11, 12, 13, 14, 15);
            acc[sub] = __builtin_amdgcn_wmma_f32_16x16x32_bf16(
                false, afrag, false, bfrag, (short)0, acc[sub], false, false);
        }
    }
    __syncthreads();                               // done reading sA

    // ---- Gamma = exp(-0.5 * d2) into LDS ----
    {
        const int n     = lane & 15;
        const int mhalf = (lane >> 4) * 8;         // C/D layout: lanes 16-31 -> M+8
        const int c     = wave * 16 + n;
        #pragma unroll
        for (int sub = 0; sub < 4; ++sub) {
            #pragma unroll
            for (int i = 0; i < 8; ++i) {
                const int r = sub * 16 + mhalf + i;
                sG[r * GSTRIDE + c] = __expf(-0.5f * acc[sub][i]);
            }
        }
    }
    __syncthreads();

    // ---- Per-row reduction: normalize, label projection, argmaxes ----
    if (tid < ROWS) {
        const int r = tid;
        const size_t b = (size_t)wgBase + r;
        const float* g = sG + r * GSTRIDE;
        float sum = 0.f, best = -1.f; int bestc = 0;
        for (int c = 0; c < CC; ++c) {
            const float v = g[c];
            sum += v;
            if (v > best) { best = v; bestc = c; }
        }
        const float inv = 1.0f / (sum + 1e-12f);
        float lbest = -1.f; int lbestk = 0;
        float* ls = out + b * KK;
        #pragma unroll
        for (int k = 0; k < KK; ++k) {
            float s = 0.f;
            const float* m = Lmask + k * CC;
            for (int c = 0; c < CC; ++c) s += g[c] * m[c];
            s *= inv;
            ls[k] = s;
            if (s > lbest) { lbest = s; lbestk = k; }
        }
        out[(size_t)BB * KK + b]      = (float)lbestk;   // preds
        out[(size_t)BB * KK + BB + b] = (float)bestc;    // clusters
    }
}

extern "C" void kernel_launch(void* const* d_in, const int* in_sizes, int n_in,
                              void* d_out, int out_size, void* d_ws, size_t ws_size,
                              hipStream_t stream) {
    const float* data  = (const float*)d_in[0];   // [B,D]
    const float* mu    = (const float*)d_in[1];   // [C,D]
    const float* S_inv = (const float*)d_in[2];   // [C,D,D]
    const int*   labels = (const int*)d_in[3];    // [C,K]

    unsigned short* Bsw = (unsigned short*)d_ws;                       // 272 KB
    float* Lmask = (float*)((char*)d_ws + (size_t)CC * KTOT * sizeof(unsigned short));

    prep_B<<<(CC * KTOT + 255) / 256, 256, 0, stream>>>(S_inv, mu, Bsw);
    prep_L<<<(KK * CC + 255) / 256, 256, 0, stream>>>(labels, Lmask);

    const size_t smem = (size_t)ROWS * ROWSTRIDE * sizeof(unsigned short); // ~137 KB
    egaussp_main<<<BB / ROWS, 256, smem, stream>>>(data, Bsw, Lmask, (float*)d_out);
}
